// UnifiedLightField_52647709114742
// MI455X (gfx1250) — matrix operand order repaired and checked by reading, not compiled, not run on hardware
//
#include <hip/hip_runtime.h>
#include <hip/hip_bf16.h>
#include <math.h>

#define N_ROWS 4096
#define D_DIM  1024
#define BK     32
#define LDSTR  40   // padded LDS row stride in ushorts (80B, keeps 16B alignment)

typedef __attribute__((ext_vector_type(16))) __bf16         v16bf;
typedef __attribute__((ext_vector_type(8)))  float          v8f;
typedef __attribute__((ext_vector_type(8)))  unsigned short us8;
typedef __attribute__((ext_vector_type(4)))  unsigned short us4;

union Frag { us8 h[2]; v16bf v; };

__device__ __forceinline__ unsigned short f32_to_bf16(float f) {
  unsigned int u = __float_as_uint(f);
  u += 0x7fffu + ((u >> 16) & 1u);      // round-to-nearest-even
  return (unsigned short)(u >> 16);
}
__device__ __forceinline__ float bf16_to_f32(unsigned short h) {
  return __uint_as_float(((unsigned int)h) << 16);
}

// ---------------------------------------------------------------- convert f32 -> bf16
__global__ __launch_bounds__(256) void cvt_bf16(const float* __restrict__ src,
                                                unsigned short* __restrict__ dst,
                                                int n4) {
  int idx = blockIdx.x * 256 + threadIdx.x;
  if (idx < n4) {
    const float4 v = ((const float4*)src)[idx];
    us4 o;
    o.x = f32_to_bf16(v.x); o.y = f32_to_bf16(v.y);
    o.z = f32_to_bf16(v.z); o.w = f32_to_bf16(v.w);
    ((us4*)dst)[idx] = o;
  }
}

// ---------------------------------------------------------------- per-row statistics
__global__ __launch_bounds__(256) void rowstats_kernel(
    const float* __restrict__ vecs, const float* __restrict__ mass,
    const float* __restrict__ hops, const int* __restrict__ alive,
    const int* __restrict__ photon, const float* __restrict__ wci,
    const float* __restrict__ wcj,  const float* __restrict__ logc,
    const float* __restrict__ logG,
    float* __restrict__ att, float* __restrict__ rep,
    float* __restrict__ sI,  float* __restrict__ sJ,
    float* __restrict__ mj,  float* __restrict__ rmask) {
  const int i = blockIdx.x;
  const int t = threadIdx.x;
  const float* row = vecs + (size_t)i * D_DIM;
  float a = 0.f, b = 0.f, n2 = 0.f;
  for (int d = t; d < D_DIM; d += 256) {
    float x = row[d];
    a += x * wci[d];
    b += x * wcj[d];
    n2 += x * x;
  }
  __shared__ float sa[256], sb[256], sn[256];
  sa[t] = a; sb[t] = b; sn[t] = n2;
  __syncthreads();
  for (int off = 128; off > 0; off >>= 1) {
    if (t < off) { sa[t] += sa[t + off]; sb[t] += sb[t + off]; sn[t] += sn[t + off]; }
    __syncthreads();
  }
  if (t == 0) {
    float c  = fmaxf(expf(logc[0]), 1e-6f);
    float G  = expf(logG[0]);
    float d  = hops[i] / c;
    float lam = photon[i] ? 1.f : expf(-d);
    float rs = fmaxf(d, 0.1f);
    float causal = (d <= 1.f) ? 1.f : expf(1.f - d);
    float ai = alive[i] ? 1.f : 0.f;
    att[i] = lam * (G * mass[i] / (rs * rs)) * causal * ai;  // attract coefficient
    rep[i] = lam * mass[i] * ai;                              // repulse coefficient
    sI[i] = sa[0];
    sJ[i] = sb[0];
    mj[i] = alive[i] ? mass[i] : 0.f;
    rmask[i] = (sqrtf(sn[0]) < 1e-8f) ? 0.f : 1.f;
  }
}

// ---------------------------------------------------------------- 128x32 tile loader (256 threads)
__device__ __forceinline__ void load_tile128x32(const unsigned short* __restrict__ g,
                                                unsigned short* lds,
                                                int rowBase, int k0, int ld, int tid) {
#pragma unroll
  for (int it = 0; it < 2; ++it) {
    int idx = tid + it * 256;                 // 0..511 chunks of 8 ushorts
    int r = idx >> 2;                         // 0..127
    int c = (idx & 3) << 3;                   // 0,8,16,24
    *(us8*)(lds + r * LDSTR + c) =
        *(const us8*)(g + (size_t)(rowBase + r) * ld + k0 + c);
  }
}

// Core: one 256-thread block computes a 128x128 tile of C = A * B^T (bf16, f32 acc).
// Wave (wr,wc) with wr=wave>>1, wc=wave&1 computes rows [wr*32,+32) x cols [wc*64,+64):
// 2 A-fragments x 4 B-fragments -> acc[2][4] of v8f.
#define GEMM_CORE(APTR, AROWBASE, BPTR, BROWBASE)                                  \
  __shared__ __attribute__((aligned(16))) unsigned short At[128 * LDSTR];          \
  __shared__ __attribute__((aligned(16))) unsigned short Bt[128 * LDSTR];          \
  const int tid  = threadIdx.x;                                                    \
  const int wave = tid >> 5, lane = tid & 31;                                      \
  const int half = lane >> 4, ln = lane & 15;                                      \
  const int wr = wave >> 1, wc = wave & 1;                                         \
  v8f zero = {};                                                                   \
  v8f acc[2][4] = {{zero, zero, zero, zero}, {zero, zero, zero, zero}};            \
  for (int k0 = 0; k0 < D_DIM; k0 += BK) {                                         \
    __syncthreads();                                                               \
    load_tile128x32(APTR, At, AROWBASE, k0, D_DIM, tid);                           \
    load_tile128x32(BPTR, Bt, BROWBASE, k0, D_DIM, tid);                           \
    __syncthreads();                                                               \
    Frag af[2];                                                                    \
    _Pragma("unroll")                                                              \
    for (int a = 0; a < 2; ++a) {                                                  \
      const unsigned short* ar = At + (wr * 32 + a * 16 + ln) * LDSTR + half * 16; \
      af[a].h[0] = *(const us8*)ar;        /* K = half*16 .. +7     */             \
      af[a].h[1] = *(const us8*)(ar + 8);  /* K = half*16+8 .. +7   */             \
    }                                                                              \
    _Pragma("unroll")                                                              \
    for (int s = 0; s < 4; ++s) {                                                  \
      Frag bfr;                                                                    \
      const unsigned short* br = Bt + (wc * 64 + s * 16 + ln) * LDSTR + half * 16; \
      bfr.h[0] = *(const us8*)br;          /* K = half*16 .. +7     */             \
      bfr.h[1] = *(const us8*)(br + 8);    /* K = half*16+8 .. +7   */             \
      _Pragma("unroll")                                                            \
      for (int a = 0; a < 2; ++a) {                                                \
        acc[a][s] = __builtin_amdgcn_wmma_f32_16x16x32_bf16(                       \
            false, af[a].v, false, bfr.v, (short)0, acc[a][s], false, false);      \
      }                                                                            \
    }                                                                              \
  }

// ---------------------------------------------------------------- GEMM1: v = vecs @ W^T + b (bf16 out)
__global__ __launch_bounds__(256) void gemm_v_kernel(
    const unsigned short* __restrict__ A,   // [N,D] bf16 vecs
    const unsigned short* __restrict__ B,   // [D,D] bf16 W (rows contracted)
    const float* __restrict__ bias,         // [D]
    unsigned short* __restrict__ V) {       // [N,D] bf16
  const int colBase = blockIdx.x * 128;
  const int rowBase = blockIdx.y * 128;
  GEMM_CORE(A, rowBase, B, colBase)
#pragma unroll
  for (int a = 0; a < 2; ++a) {
#pragma unroll
    for (int s = 0; s < 4; ++s) {
      int k = colBase + wc * 64 + s * 16 + ln;
      float bv = bias[k];
#pragma unroll
      for (int r = 0; r < 8; ++r) {
        int i = rowBase + wr * 32 + a * 16 + half * 8 + r;
        V[(size_t)i * D_DIM + k] = f32_to_bf16(acc[a][s][r] + bv);
      }
    }
  }
}

// ---------------------------------------------------------------- row norms of v
__global__ __launch_bounds__(256) void vnorm_kernel(const unsigned short* __restrict__ V,
                                                    const float* __restrict__ rmask,
                                                    float* __restrict__ inr) {
  const int i = blockIdx.x;
  const int t = threadIdx.x;
  const unsigned short* row = V + (size_t)i * D_DIM;
  float n2 = 0.f;
  for (int d = t; d < D_DIM; d += 256) {
    float x = bf16_to_f32(row[d]);
    n2 += x * x;
  }
  __shared__ float sn[256];
  sn[t] = n2;
  __syncthreads();
  for (int off = 128; off > 0; off >>= 1) {
    if (t < off) sn[t] += sn[t + off];
    __syncthreads();
  }
  if (t == 0) inr[i] = rmask[i] / fmaxf(sqrtf(sn[0]), 1e-8f);
}

// ---------------------------------------------------------------- fused GEMM2 + phi epilogue
__global__ __launch_bounds__(256) void phi_kernel(
    const unsigned short* __restrict__ V,   // [N,D] bf16
    const float* __restrict__ att, const float* __restrict__ rep,
    const float* __restrict__ sI,  const float* __restrict__ sJ,
    const float* __restrict__ mj,  const float* __restrict__ inr,
    const float* __restrict__ bconf_p,
    float* __restrict__ out) {
  const int jBase = blockIdx.x * 128;
  const int iBase = blockIdx.y * 128;
  GEMM_CORE(V, iBase, V, jBase)

  const float bconf = bconf_p[0];
#pragma unroll
  for (int a = 0; a < 2; ++a) {
    float atti[8], repi[8], sIi[8], inri[8];
#pragma unroll
    for (int r = 0; r < 8; ++r) {
      int i = iBase + wr * 32 + a * 16 + half * 8 + r;
      atti[r] = att[i]; repi[r] = rep[i]; sIi[r] = sI[i]; inri[r] = inr[i];
    }
#pragma unroll
    for (int s = 0; s < 4; ++s) {
      int j = jBase + wc * 64 + s * 16 + ln;
      float sj  = sJ[j];
      float mjj = mj[j];
      float inj = inr[j];
#pragma unroll
      for (int r = 0; r < 8; ++r) {
        int i = iBase + wr * 32 + a * 16 + half * 8 + r;
        float R = acc[a][s][r] * inri[r] * inj;                    // cosine sim (masked)
        float conf = 1.f / (1.f + expf(-(sIi[r] + sj + bconf)));   // conflict gate
        float attract = (j < i) ? atti[r] * mjj * (0.5f + 0.5f * R) : 0.f;
        float ph = (conf > 0.8f) ? (-repi[r] * conf * mjj) : attract;
        if (i == j) ph = 0.f;
        out[(size_t)i * N_ROWS + j] = ph;
      }
    }
  }
}

// ---------------------------------------------------------------- launcher
extern "C" void kernel_launch(void* const* d_in, const int* in_sizes, int n_in,
                              void* d_out, int out_size, void* d_ws, size_t ws_size,
                              hipStream_t stream) {
  const float* vecs   = (const float*)d_in[0];
  const float* mass   = (const float*)d_in[1];
  const float* hops   = (const float*)d_in[2];
  const int*   alive  = (const int*)d_in[3];
  const int*   photon = (const int*)d_in[4];
  const float* Wrw    = (const float*)d_in[5];
  const float* Wrb    = (const float*)d_in[6];
  const float* wci    = (const float*)d_in[7];
  const float* wcj    = (const float*)d_in[8];
  const float* bconf  = (const float*)d_in[9];
  const float* logc   = (const float*)d_in[10];
  const float* logG   = (const float*)d_in[11];

  char* ws = (char*)d_ws;
  unsigned short* Abf = (unsigned short*)ws;                       // 8 MB vecs bf16
  unsigned short* Wbf = (unsigned short*)(ws + (size_t)8388608);   // 2 MB W bf16
  unsigned short* Vbf = (unsigned short*)(ws + (size_t)10485760);  // 8 MB v bf16
  float* att   = (float*)(ws + (size_t)18874368);
  float* rep   = att + N_ROWS;
  float* sI    = rep + N_ROWS;
  float* sJ    = sI + N_ROWS;
  float* mjA   = sJ + N_ROWS;
  float* rmask = mjA + N_ROWS;
  float* inr   = rmask + N_ROWS;

  const int nA4 = N_ROWS * D_DIM / 4;
  const int nW4 = D_DIM * D_DIM / 4;
  cvt_bf16<<<(nA4 + 255) / 256, 256, 0, stream>>>(vecs, Abf, nA4);
  cvt_bf16<<<(nW4 + 255) / 256, 256, 0, stream>>>(Wrw, Wbf, nW4);

  rowstats_kernel<<<N_ROWS, 256, 0, stream>>>(vecs, mass, hops, alive, photon,
                                              wci, wcj, logc, logG,
                                              att, rep, sI, sJ, mjA, rmask);

  gemm_v_kernel<<<dim3(D_DIM / 128, N_ROWS / 128), 256, 0, stream>>>(Abf, Wbf, Wrb, Vbf);

  vnorm_kernel<<<N_ROWS, 256, 0, stream>>>(Vbf, rmask, inr);

  phi_kernel<<<dim3(N_ROWS / 128, N_ROWS / 128), 256, 0, stream>>>(
      Vbf, att, rep, sI, sJ, mjA, inr, bconf, (float*)d_out);
}
